// VQVAE_18279380812066
// MI455X (gfx1250) — matrix operand order repaired and checked by reading, not compiled, not run on hardware
//
#include <hip/hip_runtime.h>

// ---------------------------------------------------------------------------
// Types for CDNA5 WMMA (wave32): v16bf = 16 bf16 (8 VGPRs), v8f = 8 f32 acc.
// ---------------------------------------------------------------------------
typedef unsigned short u16;
typedef __attribute__((ext_vector_type(16))) __bf16 v16bf;
typedef __attribute__((ext_vector_type(8)))  float  v8f;

union FragAB { v16bf v; uint4 q[2]; };

__device__ __forceinline__ u16 f32_to_bf16(float x) {
  union { float f; unsigned int u; } c; c.f = x;
  unsigned int u = c.u;
  u += 0x7FFFu + ((u >> 16) & 1u);   // round to nearest even
  return (u16)(u >> 16);
}

// A fragment: 16x32 bf16 tile of row-major A[M x K].
// Lane (r, half): elems 0..7 = K(k+half*8 .. +7), elems 8..15 = K(k+16+half*8 ..)
__device__ __forceinline__ v16bf load_a_frag(const u16* __restrict__ A, int lda,
                                             int m, int k, int r, int half) {
  FragAB f;
  const u16* p = A + (size_t)(m + r) * lda + k + half * 8;
  f.q[0] = *(const uint4*)(p);
  f.q[1] = *(const uint4*)(p + 16);
  return f.v;
}

// B fragment: 32x16 bf16 tile taken from Bt[N x K] row-major (pre-transposed
// weight). Lane (r, half): column n+r, elems 0..7 = K(k+half*8 .. +7), elems 8..15 = K(k+16+half*8 ..)
__device__ __forceinline__ v16bf load_b_frag(const u16* __restrict__ Bt, int ldb,
                                             int n, int k, int r, int half) {
  FragAB f;
  const u16* p = Bt + (size_t)(n + r) * ldb + k + half * 8;
  f.q[0] = *(const uint4*)(p);
  f.q[1] = *(const uint4*)(p + 16);
  return f.v;
}

__device__ __forceinline__ v8f wmma_bf16(v16bf a, v16bf b, v8f c) {
  return __builtin_amdgcn_wmma_f32_16x16x32_bf16(false, a, false, b, (short)0,
                                                 c, false, false);
}

// ---------------------------------------------------------------------------
// Elementwise fp32 -> bf16 convert (grid-stride)
// ---------------------------------------------------------------------------
__global__ __launch_bounds__(256) void cvt_bf16(const float* __restrict__ src,
                                                u16* __restrict__ dst, size_t n) {
  size_t i = (size_t)blockIdx.x * 256 + threadIdx.x;
  size_t stride = (size_t)gridDim.x * 256;
  for (; i < n; i += stride) dst[i] = f32_to_bf16(src[i]);
}

// fp32 W[K x N] -> bf16 Wt[N x K] via 32x32 LDS tile (bank-conflict padded)
__global__ __launch_bounds__(256) void cvt_transpose(const float* __restrict__ W,
                                                     u16* __restrict__ Wt,
                                                     int K, int N) {
  __shared__ float tile[32][33];
  const int nt = N >> 5;
  const int bx = blockIdx.x % nt;     // N tile
  const int by = blockIdx.x / nt;     // K tile
  const int tx = threadIdx.x & 31;
  const int ty = threadIdx.x >> 5;    // 0..7
#pragma unroll
  for (int i = 0; i < 32; i += 8)
    tile[ty + i][tx] = W[(size_t)(by * 32 + ty + i) * N + bx * 32 + tx];
  __syncthreads();
#pragma unroll
  for (int i = 0; i < 32; i += 8)
    Wt[(size_t)(bx * 32 + ty + i) * K + by * 32 + tx] = f32_to_bf16(tile[tx][ty + i]);
}

// per-entry 0.5*||e||^2 penalty for fused argmin
__global__ __launch_bounds__(256) void cb_norm_half(const float* __restrict__ cb,
                                                    float* __restrict__ cbnh) {
  int e = blockIdx.x * 256 + threadIdx.x;
  float s = 0.f;
  for (int d = 0; d < 256; ++d) {
    float v = cb[(size_t)e * 256 + d];
    s += v * v;
  }
  cbnh[e] = 0.5f * s;
}

__global__ void zero_scalar(float* p) { *p = 0.f; }

__global__ void finalize_loss(const float* __restrict__ acc, float* __restrict__ out) {
  out[0] = 1.25f * acc[0] / (16384.f * 256.f);  // (1 + COMMIT) * mean((zq - z)^2)
}

// ---------------------------------------------------------------------------
// WMMA GEMM: C = epi(A[MxK] * Bt[NxK]^T + bias)
//   block = 256 threads = 8 waves; each wave owns a 32x64 output tile
//   (2x4 accumulator tiles, 8 v_wmma_f32_16x16x32_bf16 per 32-wide k-step).
//   Fragments are double-buffered across k-steps so the 12 b128 loads of
//   step k+1 overlap the 8 WMMAs of step k (last step peeled).
// EPI: 0 = relu -> bf16 ; 1 = identity -> f32 + bf16 ; 2 = sigmoid -> f32
// ---------------------------------------------------------------------------
template <int EPI>
__global__ __launch_bounds__(256) void gemm_wmma_bf16(
    const u16* __restrict__ A, const u16* __restrict__ Bt,
    const float* __restrict__ bias, float* __restrict__ Cf,
    u16* __restrict__ Cb, int M, int N, int K) {
  const int lane = threadIdx.x & 31;
  const int wave = threadIdx.x >> 5;
  const int r = lane & 15;
  const int half = lane >> 4;

  const int ntiles = N >> 6;               // 64-wide n strips
  const int gw = blockIdx.x * 8 + wave;
  const int mt = gw / ntiles;
  const int nt = gw - mt * ntiles;
  const int m0 = mt << 5;
  const int n0 = nt << 6;
  if (m0 >= M) return;

  const v8f vzero = {0.f, 0.f, 0.f, 0.f, 0.f, 0.f, 0.f, 0.f};
  v8f acc[2][4];
#pragma unroll
  for (int mi = 0; mi < 2; ++mi)
#pragma unroll
    for (int ni = 0; ni < 4; ++ni) acc[mi][ni] = vzero;

  // prologue: fragments for k = 0
  v16bf a0 = load_a_frag(A, K, m0, 0, r, half);
  v16bf a1 = load_a_frag(A, K, m0 + 16, 0, r, half);
  v16bf b0 = load_b_frag(Bt, K, n0, 0, r, half);
  v16bf b1 = load_b_frag(Bt, K, n0 + 16, 0, r, half);
  v16bf b2 = load_b_frag(Bt, K, n0 + 32, 0, r, half);
  v16bf b3 = load_b_frag(Bt, K, n0 + 48, 0, r, half);

  for (int k = 32; k < K; k += 32) {
    // prefetch next k-step while current WMMAs run
    v16bf na0 = load_a_frag(A, K, m0, k, r, half);
    v16bf na1 = load_a_frag(A, K, m0 + 16, k, r, half);
    v16bf nb0 = load_b_frag(Bt, K, n0, k, r, half);
    v16bf nb1 = load_b_frag(Bt, K, n0 + 16, k, r, half);
    v16bf nb2 = load_b_frag(Bt, K, n0 + 32, k, r, half);
    v16bf nb3 = load_b_frag(Bt, K, n0 + 48, k, r, half);

    acc[0][0] = wmma_bf16(a0, b0, acc[0][0]);
    acc[1][0] = wmma_bf16(a1, b0, acc[1][0]);
    acc[0][1] = wmma_bf16(a0, b1, acc[0][1]);
    acc[1][1] = wmma_bf16(a1, b1, acc[1][1]);
    acc[0][2] = wmma_bf16(a0, b2, acc[0][2]);
    acc[1][2] = wmma_bf16(a1, b2, acc[1][2]);
    acc[0][3] = wmma_bf16(a0, b3, acc[0][3]);
    acc[1][3] = wmma_bf16(a1, b3, acc[1][3]);

    a0 = na0; a1 = na1; b0 = nb0; b1 = nb1; b2 = nb2; b3 = nb3;
  }
  // epilogue k-step
  acc[0][0] = wmma_bf16(a0, b0, acc[0][0]);
  acc[1][0] = wmma_bf16(a1, b0, acc[1][0]);
  acc[0][1] = wmma_bf16(a0, b1, acc[0][1]);
  acc[1][1] = wmma_bf16(a1, b1, acc[1][1]);
  acc[0][2] = wmma_bf16(a0, b2, acc[0][2]);
  acc[1][2] = wmma_bf16(a1, b2, acc[1][2]);
  acc[0][3] = wmma_bf16(a0, b3, acc[0][3]);
  acc[1][3] = wmma_bf16(a1, b3, acc[1][3]);

  // C/D layout: VGPR j, lanes 0-15 -> row j, lanes 16-31 -> row j+8, col = r
#pragma unroll
  for (int mi = 0; mi < 2; ++mi)
#pragma unroll
    for (int ni = 0; ni < 4; ++ni) {
      const int col = n0 + ni * 16 + r;
      const float bv = bias[col];
#pragma unroll
      for (int j = 0; j < 8; ++j) {
        const int row = m0 + mi * 16 + half * 8 + j;
        float v = acc[mi][ni][j] + bv;
        const size_t o = (size_t)row * N + col;
        if (EPI == 0) {
          v = fmaxf(v, 0.f);
          Cb[o] = f32_to_bf16(v);
        } else if (EPI == 1) {
          Cf[o] = v;
          Cb[o] = f32_to_bf16(v);
        } else {
          Cf[o] = 1.f / (1.f + __expf(-v));
        }
      }
    }
}

// ---------------------------------------------------------------------------
// Fused VQ argmin:  argmin_e ||z - e||^2 == argmax_e (z.e - 0.5||e||^2).
// One wave per 16 batch rows; z rows held as 8 resident A fragments (K=256).
// Sweep the codebook 64 entries per iteration as FOUR independent WMMA
// accumulator chains: while one chain's B fragment is in flight the other
// three chains have WMMAs ready, so loads overlap matrix ops instead of the
// load -> wait(0) -> wmma lockstep a single serial chain forces.
// Final 16-lane xor-shuffle max reduction, lowest-index tie-break (matches
// jnp.argmin first-minimum semantics).
// ---------------------------------------------------------------------------
__global__ __launch_bounds__(256) void vq_argmin_wmma(
    const u16* __restrict__ Zb, const u16* __restrict__ Cbk,
    const float* __restrict__ cbnh, int* __restrict__ idx) {
  const int lane = threadIdx.x & 31;
  const int wave = threadIdx.x >> 5;
  const int r = lane & 15;
  const int half = lane >> 4;
  const int m0 = (blockIdx.x * 8 + wave) * 16;

  v16bf afr[8];
#pragma unroll
  for (int kk = 0; kk < 8; ++kk) afr[kk] = load_a_frag(Zb, 256, m0, kk * 32, r, half);

  const v8f vzero = {0.f, 0.f, 0.f, 0.f, 0.f, 0.f, 0.f, 0.f};
  float best_s[8];
  int best_i[8];
#pragma unroll
  for (int j = 0; j < 8; ++j) { best_s[j] = -3.4e38f; best_i[j] = 0; }

  for (int e0 = 0; e0 < 8192; e0 += 64) {
    v8f acc[4];
#pragma unroll
    for (int t = 0; t < 4; ++t) acc[t] = vzero;

#pragma unroll
    for (int kk = 0; kk < 8; ++kk) {
#pragma unroll
      for (int t = 0; t < 4; ++t) {
        v16bf b = load_b_frag(Cbk, 256, e0 + t * 16, kk * 32, r, half);
        acc[t] = wmma_bf16(afr[kk], b, acc[t]);   // 4 independent chains
      }
    }

#pragma unroll
    for (int t = 0; t < 4; ++t) {
      const float pen = cbnh[e0 + t * 16 + r];
#pragma unroll
      for (int j = 0; j < 8; ++j) {
        float s = acc[t][j] - pen;
        if (s > best_s[j]) { best_s[j] = s; best_i[j] = e0 + t * 16 + r; }
      }
    }
  }

#pragma unroll
  for (int off = 1; off < 16; off <<= 1) {
#pragma unroll
    for (int j = 0; j < 8; ++j) {
      float so = __shfl_xor(best_s[j], off, 32);
      int io = __shfl_xor(best_i[j], off, 32);
      if (so > best_s[j] || (so == best_s[j] && io < best_i[j])) {
        best_s[j] = so; best_i[j] = io;
      }
    }
  }
  if (r == 0) {
#pragma unroll
    for (int j = 0; j < 8; ++j) idx[m0 + half * 8 + j] = best_i[j];
  }
}

// ---------------------------------------------------------------------------
// Gather z_q = codebook[idx] (bf16 out for dec1) + accumulate sum((zq - z)^2)
// ---------------------------------------------------------------------------
__global__ __launch_bounds__(256) void gather_loss(
    const float* __restrict__ z, const float* __restrict__ cbf,
    const int* __restrict__ idx, u16* __restrict__ zqb,
    float* __restrict__ lossAcc) {
  __shared__ float red[256];
  const int b = blockIdx.x;
  const int d = threadIdx.x;              // LATENT_DIM == blockDim == 256
  const int e = idx[b];
  const float q = cbf[(size_t)e * 256 + d];
  const float zz = z[(size_t)b * 256 + d];
  const float diff = q - zz;
  zqb[(size_t)b * 256 + d] = f32_to_bf16(q);
  red[d] = diff * diff;
  __syncthreads();
  for (int s = 128; s > 0; s >>= 1) {
    if (d < s) red[d] += red[d + s];
    __syncthreads();
  }
  if (d == 0) atomicAdd(lossAcc, red[0]);
}

// ---------------------------------------------------------------------------
// Launcher
// ---------------------------------------------------------------------------
extern "C" void kernel_launch(void* const* d_in, const int* in_sizes, int n_in,
                              void* d_out, int out_size, void* d_ws, size_t ws_size,
                              hipStream_t stream) {
  (void)in_sizes; (void)n_in; (void)out_size; (void)ws_size;
  constexpr int Bm = 16384, IN = 3072, HD = 4096, LT = 256, KE = 8192;

  const float* x   = (const float*)d_in[0];
  const float* ew1 = (const float*)d_in[1];
  const float* eb1 = (const float*)d_in[2];
  const float* ew2 = (const float*)d_in[3];
  const float* eb2 = (const float*)d_in[4];
  const float* dw1 = (const float*)d_in[5];
  const float* db1 = (const float*)d_in[6];
  const float* dw2 = (const float*)d_in[7];
  const float* db2 = (const float*)d_in[8];
  const float* cb  = (const float*)d_in[9];
  float* out = (float*)d_out;

  char* wp = (char*)d_ws;
  auto carve = [&](size_t bytes) -> void* {
    void* q = (void*)wp;
    wp += (bytes + 255) & ~(size_t)255;
    return q;
  };
  u16* xb     = (u16*)carve((size_t)Bm * IN * 2);   // x bf16
  u16* w1t    = (u16*)carve((size_t)HD * IN * 2);   // enc_w1^T bf16
  u16* hb     = (u16*)carve((size_t)Bm * HD * 2);   // h bf16 (reused for h2)
  u16* w2t    = (u16*)carve((size_t)LT * HD * 2);   // enc_w2^T bf16
  float* zf   = (float*)carve((size_t)Bm * LT * 4); // z f32
  u16* zb     = (u16*)carve((size_t)Bm * LT * 2);   // z bf16
  u16* cbb    = (u16*)carve((size_t)KE * LT * 2);   // codebook bf16
  float* cbnh = (float*)carve((size_t)KE * 4);      // 0.5*||e||^2
  int* idx    = (int*)carve((size_t)Bm * 4);
  u16* zqb    = (u16*)carve((size_t)Bm * LT * 2);   // z_q bf16
  u16* w3t    = (u16*)carve((size_t)HD * LT * 2);   // dec_w1^T bf16
  u16* w4t    = (u16*)carve((size_t)IN * HD * 2);   // dec_w2^T bf16
  float* loss = (float*)carve(256);

  const dim3 blk(256);

  // --- precision conversion / weight transposition (bandwidth-trivial) ---
  cvt_bf16<<<8192, blk, 0, stream>>>(x, xb, (size_t)Bm * IN);
  cvt_transpose<<<(IN / 32) * (HD / 32), blk, 0, stream>>>(ew1, w1t, IN, HD);
  cvt_transpose<<<(HD / 32) * (LT / 32), blk, 0, stream>>>(ew2, w2t, HD, LT);
  cvt_transpose<<<(LT / 32) * (HD / 32), blk, 0, stream>>>(dw1, w3t, LT, HD);
  cvt_transpose<<<(HD / 32) * (IN / 32), blk, 0, stream>>>(dw2, w4t, HD, IN);
  cvt_bf16<<<2048, blk, 0, stream>>>(cb, cbb, (size_t)KE * LT);
  cb_norm_half<<<KE / 256, blk, 0, stream>>>(cb, cbnh);
  zero_scalar<<<1, 1, 0, stream>>>(loss);

  // --- encoder ---
  gemm_wmma_bf16<0><<<(Bm / 32) * (HD / 64) / 8, blk, 0, stream>>>(
      xb, w1t, eb1, nullptr, hb, Bm, HD, IN);                      // h = relu(x W1 + b1)
  gemm_wmma_bf16<1><<<(Bm / 32) * (LT / 64) / 8, blk, 0, stream>>>(
      hb, w2t, eb2, zf, zb, Bm, LT, HD);                           // z = h W2 + b2

  // --- vector quantizer ---
  vq_argmin_wmma<<<Bm / 16 / 8, blk, 0, stream>>>(zb, cbb, cbnh, idx);
  gather_loss<<<Bm, blk, 0, stream>>>(zf, cb, idx, zqb, loss);

  // --- decoder (hb buffer reused for h2) ---
  gemm_wmma_bf16<0><<<(Bm / 32) * (HD / 64) / 8, blk, 0, stream>>>(
      zqb, w3t, db1, nullptr, hb, Bm, HD, LT);                     // h2 = relu(zq W3 + b3)
  gemm_wmma_bf16<2><<<(Bm / 32) * (IN / 64) / 8, blk, 0, stream>>>(
      hb, w4t, db2, out, nullptr, Bm, IN, HD);                     // x_recon = sigmoid(...)

  finalize_loss<<<1, 1, 0, stream>>>(loss, out + (size_t)Bm * IN);
}
